// RNNSpatioTemporalDecoder_26027501814590
// MI455X (gfx1250) — matrix operand order repaired and checked by reading, not compiled, not run on hardware
//
#include <hip/hip_runtime.h>
#include <hip/hip_bf16.h>
#include <cstdint>

// Model dims (fixed by reference)
#define BB   8
#define NN   64
#define TT   30
#define FF   6
#define HH   256
#define KK   2
#define EE   4032          // N*(N-1)
#define DD   262           // H + F_IN
#define DP   288           // padded D_NODE (mult of 32 for WMMA K)
#define G3   800           // padded 3*D_NODE (mult of 16 for WMMA N)
#define TOUT 29            // T-1 predictions

typedef __attribute__((ext_vector_type(16))) _Float16 v16h;
typedef __attribute__((ext_vector_type(8)))  float    v8f;

union FragH { v16h v; _Float16 h[16]; unsigned u[8]; };

__device__ __forceinline__ float sigm(float x) { return 1.0f / (1.0f + expf(-x)); }

// ---------------------------------------------------------------------------
// Generic WMMA GEMM: C[M,N] = act(alpha * (A @ B) + bias)
//   A : f16 row-major [M, Kd]            (K contiguous)
//   Bt: f16 "transposed B" [N, Kd]       (K contiguous per output column)
//   C : f32 and/or f16, row-major [M, N]
// Batched over blockIdx.y with element strides sA/sB/sC/sBias.
// Requirements: M%16==0, N%16==0, Kd%32==0.
// One 16x16 output tile per wave, 8 waves per block.
// Fragment layouts per CDNA5 ISA:
//   A (16x32 f16): lane half h, dword j -> K = 16*(j/4) + 8*h + 2*(j%4) + {0,1}, M = lane%16
//   B (32x16 f16): lane half h, dword j -> K = 16*(j/4) + 8*h + 2*(j%4) + {0,1}, N = lane%16
//   C (16x16 f32): vgpr r -> M = r + 8*h, N = lane%16
// ---------------------------------------------------------------------------
__global__ __launch_bounds__(256)
void wmma_gemm_kernel(const _Float16* __restrict__ A,
                      const _Float16* __restrict__ Bt,
                      const float* __restrict__ bias,
                      float* __restrict__ Cf,
                      _Float16* __restrict__ Ch,
                      int M, int Nn, int Kd,
                      long sA, long sB, long sC, long sBias,
                      float alpha, int act)
{
    const int lane = threadIdx.x & 31;
    const int wid  = threadIdx.x >> 5;
    const int tilesN = Nn >> 4;
    const int tilesM = M >> 4;
    const int tile = (int)blockIdx.x * 8 + wid;
    if (tile >= tilesM * tilesN) return;          // wave-uniform exit (EXEC stays full)
    const int tm = tile / tilesN;
    const int tn = tile % tilesN;
    const int z  = blockIdx.y;

    const _Float16* Az  = A  + (size_t)z * sA;
    const _Float16* Btz = Bt + (size_t)z * sB;
    const float*    bz  = bias ? (bias + (size_t)z * sBias) : nullptr;

    const int h  = lane >> 4;   // lane half
    const int mn = lane & 15;   // M index for A, N index for B/C

    const _Float16* arow = Az  + (size_t)(tm * 16 + mn) * Kd;
    const _Float16* brow = Btz + (size_t)(tn * 16 + mn) * Kd;

    v8f acc = {0.f, 0.f, 0.f, 0.f, 0.f, 0.f, 0.f, 0.f};

    for (int k0 = 0; k0 < Kd; k0 += 32) {
        FragH fa, fb;
#pragma unroll
        for (int j = 0; j < 8; ++j) {
            const int ka = k0 + ((j >> 2) << 4) + (h << 3) + ((j & 3) << 1);
            fa.u[j] = *(const unsigned*)(arow + ka);
            const int kb = k0 + ((j >> 2) << 4) + (h << 3) + ((j & 3) << 1);
            fb.u[j] = *(const unsigned*)(brow + kb);
        }
        acc = __builtin_amdgcn_wmma_f32_16x16x32_f16(
            /*neg_a=*/false, fa.v, /*neg_b=*/false, fb.v,
            /*c_mod=*/(short)0, acc, /*reuse_a=*/false, /*reuse_b=*/false);
    }

#pragma unroll
    for (int r = 0; r < 8; ++r) {
        const int row = tm * 16 + r + 8 * h;
        const int col = tn * 16 + mn;
        float v = acc[r] * alpha;
        if (bz) v += bz[col];
        if (act == 1)      v = tanhf(v);
        else if (act == 2) v = fmaxf(v, 0.0f);
        const size_t ci = (size_t)z * sC + (size_t)row * Nn + col;
        if (Cf) Cf[ci] = v;
        if (Ch) Ch[ci] = (_Float16)v;
    }
}

// f32 weight [nb][Ks][Ns] -> padded f16 transposed [nb][Nd][Kd] (zero pad)
__global__ void pad_weight_kernel(const float* __restrict__ src,
                                  _Float16* __restrict__ dst,
                                  int Ks, int Ns, int Kd, int Nd,
                                  long sSrc, int nb)
{
    long i = (long)blockIdx.x * blockDim.x + threadIdx.x;
    const long tot = (long)nb * Nd * Kd;
    if (i >= tot) return;
    const int k = (int)(i % Kd);
    long r = i / Kd;
    const int n = (int)(r % Nd);
    const int z = (int)(r / Nd);
    float v = (k < Ks && n < Ns) ? src[(size_t)z * sSrc + (size_t)k * Ns + n] : 0.0f;
    dst[i] = (_Float16)v;
}

// pre_msg[e, 0:6]=send, [6:12]=recv, [12:32]=0   (single batch b, time t)
__global__ void pre_msg_kernel(const float* __restrict__ data,
                               const float* __restrict__ rel_rec,
                               const float* __restrict__ rel_send,
                               _Float16* __restrict__ pm, int b, int t)
{
    const int e = blockIdx.x * blockDim.x + threadIdx.x;
    if (e >= EE) return;
    const float* rr = rel_rec  + (((size_t)b * TT + t) * EE + e) * NN;
    const float* rs = rel_send + (((size_t)b * TT + t) * EE + e) * NN;
    float sd[FF] = {0.f, 0.f, 0.f, 0.f, 0.f, 0.f};
    float rc[FF] = {0.f, 0.f, 0.f, 0.f, 0.f, 0.f};
    for (int n = 0; n < NN; ++n) {
        const float* x = data + ((size_t)(b * NN + n) * TT + t) * FF;
        const float a = rr[n];
        const float s = rs[n];
#pragma unroll
        for (int f = 0; f < FF; ++f) { const float v = x[f]; rc[f] += a * v; sd[f] += s * v; }
    }
    _Float16* o = pm + (size_t)e * 32;
#pragma unroll
    for (int f = 0; f < FF; ++f) { o[f] = (_Float16)sd[f]; o[FF + f] = (_Float16)rc[f]; }
#pragma unroll
    for (int c = 2 * FF; c < 32; ++c) o[c] = (_Float16)0.0f;
}

// all_msgs = 0.5*(tanh_m2[k=0]*rt0 + tanh_m2[k=1]*rt1)   (single batch)
__global__ void combine_kernel(const float* __restrict__ m2,
                               const float* __restrict__ rt,      // rel_type + b*E*K
                               float* __restrict__ f32out,        // nullable
                               _Float16* __restrict__ h16out,
                               _Float16* __restrict__ hTout)      // nullable [H][E]
{
    const int i = blockIdx.x * blockDim.x + threadIdx.x;
    if (i >= EE * HH) return;
    const int e = i >> 8;
    const int c = i & 255;
    const float v = 0.5f * (m2[i] * rt[e * KK] + m2[(size_t)EE * HH + i] * rt[e * KK + 1]);
    if (f32out) f32out[i] = v;
    h16out[i] = (_Float16)v;
    if (hTout) hTout[(size_t)c * EE + e] = (_Float16)v;
}

// edge GRU elementwise (single batch); h buffers pre-offset to batch b
__global__ void gru_edge_kernel(const float* __restrict__ gi,
                                const float* __restrict__ gh,
                                const float* __restrict__ bih,
                                const float* __restrict__ bhh,
                                float* __restrict__ h32,
                                _Float16* __restrict__ h16,
                                _Float16* __restrict__ hT)
{
    const int i = blockIdx.x * blockDim.x + threadIdx.x;
    if (i >= EE * HH) return;
    const int row = i >> 8;
    const int c   = i & 255;
    const size_t g = (size_t)row * 3 * HH;
    const float ir = gi[g + c] + bih[c];
    const float iz = gi[g + HH + c] + bih[HH + c];
    const float in_ = gi[g + 2 * HH + c] + bih[2 * HH + c];
    const float hr = gh[g + c] + bhh[c];
    const float hz = gh[g + HH + c] + bhh[HH + c];
    const float hn = gh[g + 2 * HH + c] + bhh[2 * HH + c];
    const float r = sigm(ir + hr);
    const float z = sigm(iz + hz);
    const float n = tanhf(in_ + r * hn);
    const float ho = h32[i];
    const float v = (1.0f - z) * n + z * ho;
    h32[i] = v;
    h16[i] = (_Float16)v;
    hT[(size_t)c * EE + row] = (_Float16)v;
}

// rrT[b][n][e] = rel_rec[b][t][e][n] as f16 (A matrix of the agg GEMM)
__global__ void transpose_rr_kernel(const float* __restrict__ rel_rec,
                                    _Float16* __restrict__ rrT, int t)
{
    long i = (long)blockIdx.x * blockDim.x + threadIdx.x;
    if (i >= (long)BB * NN * EE) return;
    const int e = (int)(i % EE);
    long r = i / EE;
    const int n = (int)(r % NN);
    const int b = (int)(r / NN);
    rrT[i] = (_Float16)rel_rec[(((size_t)b * TT + t) * EE + e) * NN + n];
}

// cat[row, 0:6]=x, [6:262]=agg (already /6), [262:288]=0
__global__ void build_cat_kernel(const float* __restrict__ data,
                                 const float* __restrict__ agg,
                                 float* __restrict__ cf,
                                 _Float16* __restrict__ ch, int t)
{
    const int i = blockIdx.x * blockDim.x + threadIdx.x;
    if (i >= BB * NN * DP) return;
    const int c = i % DP;
    const int row = i / DP;
    float v = 0.0f;
    if (c < FF)            v = data[((size_t)row * TT + t) * FF + c];
    else if (c < FF + HH)  v = agg[(size_t)row * HH + (c - FF)];
    cf[i] = v;
    ch[i] = (_Float16)v;
}

// node GRU elementwise; gate rows stride G3, gate cols split at DD
__global__ void gru_node_kernel(const float* __restrict__ gi,
                                const float* __restrict__ gh,
                                const float* __restrict__ bih,
                                const float* __restrict__ bhh,
                                float* __restrict__ h32,
                                _Float16* __restrict__ h16)
{
    const int i = blockIdx.x * blockDim.x + threadIdx.x;
    if (i >= BB * NN * DD) return;
    const int d = i % DD;
    const int row = i / DD;
    const size_t g = (size_t)row * G3;
    const float ir = gi[g + d] + bih[d];
    const float iz = gi[g + DD + d] + bih[DD + d];
    const float in_ = gi[g + 2 * DD + d] + bih[2 * DD + d];
    const float hr = gh[g + d] + bhh[d];
    const float hz = gh[g + DD + d] + bhh[DD + d];
    const float hn = gh[g + 2 * DD + d] + bhh[2 * DD + d];
    const float r = sigm(ir + hr);
    const float z = sigm(iz + hz);
    const float n = tanhf(in_ + r * hn);
    const size_t hi = (size_t)row * DP + d;
    const float ho = h32[hi];
    const float v = (1.0f - z) * n + z * ho;
    h32[hi] = v;
    h16[hi] = (_Float16)v;
}

// pred = x + h2 @ Wo3 + bo3 ; scatter to out[b][n][t][f]
__global__ void final_out_kernel(const float* __restrict__ h2,
                                 const float* __restrict__ Wo3,
                                 const float* __restrict__ bo3,
                                 const float* __restrict__ data,
                                 float* __restrict__ out, int t)
{
    const int i = blockIdx.x * blockDim.x + threadIdx.x;
    if (i >= BB * NN * FF) return;
    const int f = i % FF;
    const int row = i / FF;
    float acc = bo3[f];
    const float* hrow = h2 + (size_t)row * HH;
    for (int c = 0; c < HH; ++c) acc += hrow[c] * Wo3[c * FF + f];
    const float x = data[((size_t)row * TT + t) * FF + f];
    out[((size_t)row * TOUT + t) * FF + f] = x + acc;
}

// ---------------------------------------------------------------------------
static inline void* carve(char*& p, size_t bytes)
{
    void* r = (void*)p;
    p += (bytes + 255) & ~(size_t)255;
    return r;
}

extern "C" void kernel_launch(void* const* d_in, const int* in_sizes, int n_in,
                              void* d_out, int out_size, void* d_ws, size_t ws_size,
                              hipStream_t stream)
{
    (void)in_sizes; (void)n_in; (void)out_size; (void)ws_size;
    const float* data     = (const float*)d_in[0];
    const float* rel_type = (const float*)d_in[1];
    const float* rel_rec  = (const float*)d_in[2];
    const float* rel_send = (const float*)d_in[3];
    const float* W1    = (const float*)d_in[4];
    const float* b1    = (const float*)d_in[5];
    const float* W2    = (const float*)d_in[6];
    const float* b2    = (const float*)d_in[7];
    const float* We_ih = (const float*)d_in[8];
    const float* We_hh = (const float*)d_in[9];
    const float* be_ih = (const float*)d_in[10];
    const float* be_hh = (const float*)d_in[11];
    const float* Wn_ih = (const float*)d_in[12];
    const float* Wn_hh = (const float*)d_in[13];
    const float* bn_ih = (const float*)d_in[14];
    const float* bn_hh = (const float*)d_in[15];
    const float* Wo1   = (const float*)d_in[16];
    const float* bo1   = (const float*)d_in[17];
    const float* Wo2   = (const float*)d_in[18];
    const float* bo2   = (const float*)d_in[19];
    const float* Wo3   = (const float*)d_in[20];
    const float* bo3   = (const float*)d_in[21];
    float* out = (float*)d_out;

    char* p = (char*)d_ws;
    float*    h_edge32  = (float*)   carve(p, (size_t)BB * EE * HH * sizeof(float));
    _Float16* h_edge16  = (_Float16*)carve(p, (size_t)BB * EE * HH * sizeof(_Float16));
    _Float16* h_edge16T = (_Float16*)carve(p, (size_t)BB * HH * EE * sizeof(_Float16));
    float*    h_node32  = (float*)   carve(p, (size_t)BB * NN * DP * sizeof(float));
    _Float16* h_node16  = (_Float16*)carve(p, (size_t)BB * NN * DP * sizeof(_Float16));
    _Float16* pm16      = (_Float16*)carve(p, (size_t)EE * 32 * sizeof(_Float16));
    _Float16* m1h       = (_Float16*)carve(p, (size_t)KK * EE * HH * sizeof(_Float16));
    float*    m2f       = (float*)   carve(p, (size_t)KK * EE * HH * sizeof(float));
    _Float16* amsg16    = (_Float16*)carve(p, (size_t)EE * HH * sizeof(_Float16));
    float*    gi_b      = (float*)   carve(p, (size_t)EE * 3 * HH * sizeof(float));
    float*    gh_b      = (float*)   carve(p, (size_t)EE * 3 * HH * sizeof(float));
    _Float16* rrT       = (_Float16*)carve(p, (size_t)BB * NN * EE * sizeof(_Float16));
    float*    agg       = (float*)   carve(p, (size_t)BB * NN * HH * sizeof(float));
    float*    catf      = (float*)   carve(p, (size_t)BB * NN * DP * sizeof(float));
    _Float16* cath      = (_Float16*)carve(p, (size_t)BB * NN * DP * sizeof(_Float16));
    float*    gin       = (float*)   carve(p, (size_t)BB * NN * G3 * sizeof(float));
    float*    ghn       = (float*)   carve(p, (size_t)BB * NN * G3 * sizeof(float));
    _Float16* h1h       = (_Float16*)carve(p, (size_t)BB * NN * HH * sizeof(_Float16));
    float*    h2f       = (float*)   carve(p, (size_t)BB * NN * HH * sizeof(float));
    _Float16* W1h       = (_Float16*)carve(p, (size_t)KK * HH * 32 * sizeof(_Float16));
    _Float16* W2h       = (_Float16*)carve(p, (size_t)KK * HH * HH * sizeof(_Float16));
    _Float16* WeihH     = (_Float16*)carve(p, (size_t)3 * HH * HH * sizeof(_Float16));
    _Float16* WehhH     = (_Float16*)carve(p, (size_t)3 * HH * HH * sizeof(_Float16));
    _Float16* WnihH     = (_Float16*)carve(p, (size_t)G3 * DP * sizeof(_Float16));
    _Float16* WnhhH     = (_Float16*)carve(p, (size_t)G3 * DP * sizeof(_Float16));
    _Float16* Wo1H      = (_Float16*)carve(p, (size_t)HH * DP * sizeof(_Float16));
    _Float16* Wo2H      = (_Float16*)carve(p, (size_t)HH * HH * sizeof(_Float16));

    auto gemm = [&](const _Float16* A, const _Float16* Bt, const float* bias,
                    float* Cf, _Float16* Ch, int M, int Nn, int Kd,
                    long sA, long sB, long sC, long sBias,
                    float alpha, int act, int nb) {
        const int tiles = (M / 16) * (Nn / 16);
        dim3 grid((unsigned)((tiles + 7) / 8), (unsigned)nb, 1);
        wmma_gemm_kernel<<<grid, dim3(256), 0, stream>>>(
            A, Bt, bias, Cf, Ch, M, Nn, Kd, sA, sB, sC, sBias, alpha, act);
    };
    auto padw = [&](const float* src, _Float16* dst, int Ks, int Ns, int Kd, int Nd,
                    long sSrc, int nb) {
        const long tot = (long)nb * Nd * Kd;
        pad_weight_kernel<<<(unsigned)((tot + 255) / 256), 256, 0, stream>>>(
            src, dst, Ks, Ns, Kd, Nd, sSrc, nb);
    };

    // --- one-time weight conversion (f32 -> f16, transposed [N][K], zero padded)
    padw(W1,    W1h,   2 * FF, HH,     32, HH,     (long)2 * FF * HH, KK);
    padw(W2,    W2h,   HH,     HH,     HH, HH,     (long)HH * HH,     KK);
    padw(We_ih, WeihH, HH,     3 * HH, HH, 3 * HH, 0, 1);
    padw(We_hh, WehhH, HH,     3 * HH, HH, 3 * HH, 0, 1);
    padw(Wn_ih, WnihH, DD,     3 * DD, DP, G3,     0, 1);
    padw(Wn_hh, WnhhH, DD,     3 * DD, DP, G3,     0, 1);
    padw(Wo1,   Wo1H,  DD,     HH,     DP, HH,     0, 1);
    padw(Wo2,   Wo2H,  HH,     HH,     HH, HH,     0, 1);

    for (int t = 0; t < TOUT; ++t) {
        const bool first = (t == 0);
        transpose_rr_kernel<<<(BB * NN * EE + 255) / 256, 256, 0, stream>>>(rel_rec, rrT, t);

        for (int b = 0; b < BB; ++b) {
            pre_msg_kernel<<<(EE + 255) / 256, 256, 0, stream>>>(
                data, rel_rec, rel_send, pm16, b, t);
            // m1 = tanh(pm @ W1[k] + b1[k])  -> f16, batched over k
            gemm(pm16, W1h, b1, nullptr, m1h, EE, HH, 32,
                 0, (long)32 * HH, (long)EE * HH, HH, 1.0f, 1, KK);
            // m2 = tanh(m1 @ W2[k] + b2[k])  -> f32
            gemm(m1h, W2h, b2, m2f, nullptr, EE, HH, HH,
                 (long)EE * HH, (long)HH * HH, (long)EE * HH, HH, 1.0f, 1, KK);
            if (first) {
                combine_kernel<<<(EE * HH + 255) / 256, 256, 0, stream>>>(
                    m2f, rel_type + (size_t)b * EE * KK,
                    h_edge32 + (size_t)b * EE * HH,
                    h_edge16 + (size_t)b * EE * HH,
                    h_edge16T + (size_t)b * HH * EE);
            } else {
                combine_kernel<<<(EE * HH + 255) / 256, 256, 0, stream>>>(
                    m2f, rel_type + (size_t)b * EE * KK, nullptr, amsg16, nullptr);
                // gi = all_msgs @ We_ih ; gh = h_edge @ We_hh (biases in elementwise)
                gemm(amsg16, WeihH, nullptr, gi_b, nullptr, EE, 3 * HH, HH,
                     0, 0, 0, 0, 1.0f, 0, 1);
                gemm(h_edge16 + (size_t)b * EE * HH, WehhH, nullptr, gh_b, nullptr,
                     EE, 3 * HH, HH, 0, 0, 0, 0, 1.0f, 0, 1);
                gru_edge_kernel<<<(EE * HH + 255) / 256, 256, 0, stream>>>(
                    gi_b, gh_b, be_ih, be_hh,
                    h_edge32 + (size_t)b * EE * HH,
                    h_edge16 + (size_t)b * EE * HH,
                    h_edge16T + (size_t)b * HH * EE);
            }
        }

        // agg[b] = (rrT[b] @ msgs[b]) / 6, batched over b
        gemm(rrT, h_edge16T, nullptr, agg, nullptr, NN, HH, EE,
             (long)NN * EE, (long)HH * EE, (long)NN * HH, 0, 1.0f / 6.0f, 0, BB);

        if (first) {
            build_cat_kernel<<<(BB * NN * DP + 255) / 256, 256, 0, stream>>>(
                data, agg, h_node32, h_node16, t);
        } else {
            build_cat_kernel<<<(BB * NN * DP + 255) / 256, 256, 0, stream>>>(
                data, agg, catf, cath, t);
            gemm(cath, WnihH, nullptr, gin, nullptr, BB * NN, G3, DP,
                 0, 0, 0, 0, 1.0f, 0, 1);
            gemm(h_node16, WnhhH, nullptr, ghn, nullptr, BB * NN, G3, DP,
                 0, 0, 0, 0, 1.0f, 0, 1);
            gru_node_kernel<<<(BB * NN * DD + 255) / 256, 256, 0, stream>>>(
                gin, ghn, bn_ih, bn_hh, h_node32, h_node16);
        }

        // output MLP: h1 = relu(h_node @ Wo1 + bo1); h2 = relu(h1 @ Wo2 + bo2)
        gemm(h_node16, Wo1H, bo1, nullptr, h1h, BB * NN, HH, DP,
             0, 0, 0, 0, 1.0f, 2, 1);
        gemm(h1h, Wo2H, bo2, h2f, nullptr, BB * NN, HH, HH,
             0, 0, 0, 0, 1.0f, 2, 1);
        final_out_kernel<<<(BB * NN * FF + 255) / 256, 256, 0, stream>>>(
            h2f, Wo3, bo3, data, out, t);
    }
}
